// SwinTransformerBlock3D_27324581937423
// MI455X (gfx1250) — matrix operand order repaired and checked by reading, not compiled, not run on hardware
//
#include <hip/hip_runtime.h>
#include <hip/hip_bf16.h>

typedef __attribute__((ext_vector_type(16))) _Float16 v16h;
typedef __attribute__((ext_vector_type(8)))  _Float16 v8h;
typedef __attribute__((ext_vector_type(8)))  float    v8f;

#define WIN3      343
#define MPAD      352        // 22 * 16
#define MT        22
#define NWINDOWS  512        // 2 * 8 * 8 * 4
#define NWH       2048       // windows * heads
#define EMBED     96
#define HEADS     4
#define HDIM      24
#define HPAD      32
#define TOKENS    175616     // 2*56*56*28  (== 512*343)
#define HIDDEN    384
#define QSCALE    0.2041241452319315f   // 24^-0.5

// ---------------- workspace layout (bytes) ----------------
// weights (f16, transposed BT[n][k])
static constexpr size_t OFF_WQKVT = 0;                                // 288*96*2 = 55296
static constexpr size_t OFF_WOUTT = OFF_WQKVT + 288 * 96 * 2;         // 96*96*2  = 18432
static constexpr size_t OFF_WM1T  = OFF_WOUTT + 96 * 96 * 2;          // 384*96*2 = 73728
static constexpr size_t OFF_WM2T  = OFF_WM1T + 384 * 96 * 2;          // 96*384*2 = 73728
static constexpr size_t OFF_HW    = OFF_WM2T + 96 * 384 * 2;          // 221184
static constexpr size_t SZ_HW     = (size_t)NWINDOWS * MPAD * EMBED * 2;   // 34.6 MB (also o_buf)
static constexpr size_t OFF_Q     = OFF_HW + SZ_HW;
static constexpr size_t SZ_QKV    = (size_t)NWH * MPAD * HPAD * 2;         // 46.1 MB each
static constexpr size_t OFF_K     = OFF_Q + SZ_QKV;
static constexpr size_t OFF_VT    = OFF_K + SZ_QKV;
static constexpr size_t OFF_X2    = OFF_VT + SZ_QKV;                       // f32 attn output+residual
static constexpr size_t SZ_X2     = (size_t)TOKENS * EMBED * 4;            // 67.4 MB
static constexpr size_t OFF_H1    = OFF_X2 + SZ_X2;                        // f16 [TOKENS][384]
static constexpr size_t OFF_M16   = OFF_Q;                                 // reuse q after attention

// ---------------- fragment helpers (contiguous k order, same for A and B) ----------------
union U16H { v16h v; v8h h[2]; };

static __device__ __forceinline__ v8f vzero() {
    v8f z = {0.f, 0.f, 0.f, 0.f, 0.f, 0.f, 0.f, 0.f};
    return z;
}

// A (16x32 f16): lane holds row m=lane&15; element e -> K = e + 16*(lane>=16)
static __device__ __forceinline__ v16h load_a_frag(const _Float16* base, int ld) {
    int lane = threadIdx.x & 31;
    int m = lane & 15, hi = lane >> 4;
    const _Float16* p = base + m * ld + hi * 16;
    U16H u;
    u.h[0] = *(const v8h*)p;
    u.h[1] = *(const v8h*)(p + 8);
    return u.v;
}

// B (32x16 f16) from transposed BT[n][k] (row stride ldk in halfs), tile origin passed in.
// element e -> K = e + 16*(lane>=16), N = lane&15   => one contiguous 32B load per lane
static __device__ __forceinline__ v16h load_b_frag(const _Float16* bt_tile, int ldk) {
    int lane = threadIdx.x & 31;
    const _Float16* p = bt_tile + (lane & 15) * ldk + (lane >> 4) * 16;
    return *(const v16h*)p;
}

static __device__ __forceinline__ v8f wmma16(v16h a, v16h b, v8f c) {
    return __builtin_amdgcn_wmma_f32_16x16x32_f16(false, a, false, b, (short)0, c, false, false);
}

static __device__ __forceinline__ float wred_sum(float v) {
    for (int m = 16; m >= 1; m >>= 1) v += __shfl_xor(v, m, 32);
    return v;
}
static __device__ __forceinline__ float wred_max(float v) {
    for (int m = 16; m >= 1; m >>= 1) v = fmaxf(v, __shfl_xor(v, m, 32));
    return v;
}

// ---------------- kernel 1: weight convert + transpose to f16 ----------------
__global__ void k_cvt_weights(const float* wqkv, const float* wout, const float* wm1, const float* wm2,
                              _Float16* wqkvT, _Float16* woutT, _Float16* wm1T, _Float16* wm2T) {
    int t = blockIdx.x * blockDim.x + threadIdx.x;
    if (t < 288 * 96) { int n = t / 96, k = t % 96;  wqkvT[t] = (_Float16)wqkv[k * 288 + n]; }
    if (t < 96 * 96)  { int n = t / 96, k = t % 96;  woutT[t] = (_Float16)wout[k * 96 + n]; }
    if (t < 384 * 96) { int n = t / 96, k = t % 96;  wm1T[t]  = (_Float16)wm1[k * 384 + n]; }
    if (t < 96 * 384) { int n = t / 384, k = t % 384; wm2T[t] = (_Float16)wm2[k * 96 + n]; }
}

// ---------------- kernel 2: LN1 + roll(-3) + window partition -> hw f16 ----------------
__global__ void k_ln1_partition(const float* __restrict__ x, const float* __restrict__ g,
                                const float* __restrict__ b, _Float16* __restrict__ hw) {
    int token = blockIdx.x * 8 + (threadIdx.x >> 5);
    if (token >= TOKENS) return;
    int lane = threadIdx.x & 31;
    int wi = token / WIN3, n = token % WIN3;
    int bb = wi >> 8, rem = wi & 255;
    int nh = rem >> 5, nw = (rem >> 2) & 7, nd = rem & 3;
    int i = n / 49, j = (n / 7) % 7, k = n % 7;
    int hr = nh * 7 + i, wr = nw * 7 + j, dr = nd * 7 + k;   // rolled coords
    int hs = (hr + 3) % 56, wsrc = (wr + 3) % 56, ds = (dr + 3) % 28;
    const float* src = x + ((((size_t)bb * 56 + hs) * 56 + wsrc) * 28 + ds) * EMBED;
    float v0 = src[lane], v1 = src[lane + 32], v2 = src[lane + 64];
    float mu = wred_sum(v0 + v1 + v2) * (1.f / 96.f);
    float d0 = v0 - mu, d1 = v1 - mu, d2 = v2 - mu;
    float var = wred_sum(d0 * d0 + d1 * d1 + d2 * d2) * (1.f / 96.f);
    float rs = rsqrtf(var + 1e-5f);
    _Float16* dst = hw + ((size_t)wi * MPAD + n) * EMBED;
    dst[lane]      = (_Float16)(d0 * rs * g[lane]      + b[lane]);
    dst[lane + 32] = (_Float16)(d1 * rs * g[lane + 32] + b[lane + 32]);
    dst[lane + 64] = (_Float16)(d2 * rs * g[lane + 64] + b[lane + 64]);
}

// ---------------- kernel 3: QKV GEMM (per window m-tile, z chunks q/k/v) ----------------
__global__ void k_qkv(const _Float16* __restrict__ hw, const _Float16* __restrict__ wqkvT,
                      const float* __restrict__ bqkv,
                      _Float16* __restrict__ q, _Float16* __restrict__ kk, _Float16* __restrict__ vT) {
    int mt = blockIdx.x, wi = blockIdx.y, z = blockIdx.z;
    int lane = threadIdx.x, l15 = lane & 15, hi = lane >> 4;
    v8f acc[6];
#pragma unroll
    for (int nt = 0; nt < 6; ++nt) acc[nt] = vzero();
    const _Float16* A = hw + ((size_t)wi * MPAD + mt * 16) * EMBED;
#pragma unroll
    for (int kt = 0; kt < 3; ++kt) {
        v16h a = load_a_frag(A + kt * 32, EMBED);
#pragma unroll
        for (int nt = 0; nt < 6; ++nt) {
            v16h bfr = load_b_frag(wqkvT + (size_t)(z * 96 + nt * 16) * EMBED + kt * 32, EMBED);
            acc[nt] = wmma16(a, bfr, acc[nt]);
        }
    }
#pragma unroll
    for (int nt = 0; nt < 6; ++nt) {
        int cc = nt * 16 + l15;                 // 0..95 within chunk
        int head = cc / 24, dim = cc % 24;
        float bias = bqkv[z * 96 + cc];
        int wh = wi * 4 + head;
#pragma unroll
        for (int r = 0; r < 8; ++r) {
            int row = mt * 16 + r + 8 * hi;
            float val = acc[nt][r] + bias;
            if (z == 0)      q [((size_t)wh * MPAD + row) * HPAD + dim] = (_Float16)(val * QSCALE);
            else if (z == 1) kk[((size_t)wh * MPAD + row) * HPAD + dim] = (_Float16)val;
            else             vT[((size_t)wh * HPAD + dim) * MPAD + row] = (_Float16)val;
        }
    }
}

// ---------------- kernel 4: windowed attention (per window-head, 16-row tile) ----------------
__global__ void k_attn(const _Float16* __restrict__ q, const _Float16* __restrict__ kk,
                       const _Float16* __restrict__ vT, _Float16* __restrict__ obuf) {
    __shared__ __align__(32) float    S[16][360];
    __shared__ __align__(32) _Float16 P[16][352];
    __shared__ float rowsum[16];
    int mt = blockIdx.x, wh = blockIdx.y;
    int lane = threadIdx.x, l15 = lane & 15, hi = lane >> 4;

    // S = q @ k^T  (head dim padded to one 32-wide K tile; scale folded into q)
    v16h aq = load_a_frag(q + ((size_t)wh * MPAD + mt * 16) * HPAD, HPAD);
#pragma unroll
    for (int nt = 0; nt < MT; ++nt) {
        v16h bk = load_b_frag(kk + ((size_t)wh * MPAD + nt * 16) * HPAD, HPAD);
        v8f s = vzero();
        s = wmma16(aq, bk, s);
#pragma unroll
        for (int r = 0; r < 8; ++r) S[r + 8 * hi][nt * 16 + l15] = s[r];
    }
    __syncthreads();

    // masked softmax (valid keys: 0..342); keep row sums for late normalization
    for (int r = 0; r < 16; ++r) {
        float mx = -3.0e38f;
        for (int c = lane; c < WIN3; c += 32) mx = fmaxf(mx, S[r][c]);
        mx = wred_max(mx);
        float sm = 0.f;
        for (int c = lane; c < MPAD; c += 32) {
            float e = (c < WIN3) ? __expf(S[r][c] - mx) : 0.f;
            P[r][c] = (_Float16)e;
            sm += e;
        }
        sm = wred_sum(sm);
        if (lane == 0) rowsum[r] = sm;
    }
    __syncthreads();

    // O = P @ V   (vT is [dim][key] so B frags are contiguous 32B loads)
    v8f o0 = vzero(), o1 = vzero();
    const _Float16* vb = vT + (size_t)wh * HPAD * MPAD;
#pragma unroll
    for (int kt = 0; kt < 11; ++kt) {
        v16h ap = load_a_frag(&P[0][0] + kt * 32, 352);
        v16h b0 = load_b_frag(vb + kt * 32, MPAD);                 // dims 0..15
        v16h b1 = load_b_frag(vb + 16 * MPAD + kt * 32, MPAD);     // dims 16..31 (24..31 zero)
        o0 = wmma16(ap, b0, o0);
        o1 = wmma16(ap, b1, o1);
    }
    int wi = wh >> 2, head = wh & 3;
#pragma unroll
    for (int r = 0; r < 8; ++r) {
        int row = mt * 16 + r + 8 * hi;
        float inv = 1.f / rowsum[r + 8 * hi];
        _Float16* od = obuf + ((size_t)wi * MPAD + row) * EMBED + head * HDIM;
        od[l15] = (_Float16)(o0[r] * inv);
        if (l15 < 8) od[16 + l15] = (_Float16)(o1[r] * inv);
    }
}

// ---------------- kernel 5: out-proj + window reverse + roll(+3) + residual ----------------
__global__ void k_proj(const _Float16* __restrict__ obuf, const _Float16* __restrict__ woutT,
                       const float* __restrict__ bout, const float* __restrict__ xin,
                       float* __restrict__ x2) {
    int mt = blockIdx.x, wi = blockIdx.y;
    int lane = threadIdx.x, l15 = lane & 15, hi = lane >> 4;
    v8f acc[6];
#pragma unroll
    for (int nt = 0; nt < 6; ++nt) acc[nt] = vzero();
    const _Float16* A = obuf + ((size_t)wi * MPAD + mt * 16) * EMBED;
#pragma unroll
    for (int kt = 0; kt < 3; ++kt) {
        v16h a = load_a_frag(A + kt * 32, EMBED);
#pragma unroll
        for (int nt = 0; nt < 6; ++nt) {
            v16h bfr = load_b_frag(woutT + (size_t)(nt * 16) * EMBED + kt * 32, EMBED);
            acc[nt] = wmma16(a, bfr, acc[nt]);
        }
    }
    int bb = wi >> 8, rem = wi & 255;
    int nh = rem >> 5, nw = (rem >> 2) & 7, nd = rem & 3;
#pragma unroll
    for (int r = 0; r < 8; ++r) {
        int row = mt * 16 + r + 8 * hi;
        if (row >= WIN3) continue;
        int i = row / 49, j = (row / 7) % 7, k = row % 7;
        int hr = nh * 7 + i, wr = nw * 7 + j, dr = nd * 7 + k;   // rolled coords
        int hd = (hr + 3) % 56, wd = (wr + 3) % 56, dd = (dr + 3) % 28;  // un-rolled dest
        size_t base = ((((size_t)bb * 56 + hd) * 56 + wd) * 28 + dd) * EMBED;
#pragma unroll
        for (int nt = 0; nt < 6; ++nt) {
            int c = nt * 16 + l15;
            x2[base + c] = xin[base + c] + acc[nt][r] + bout[c];
        }
    }
}

// ---------------- kernel 6: LN2 -> f16 ----------------
__global__ void k_ln2(const float* __restrict__ x2, const float* __restrict__ g,
                      const float* __restrict__ b, _Float16* __restrict__ m16) {
    int token = blockIdx.x * 8 + (threadIdx.x >> 5);
    if (token >= TOKENS) return;
    int lane = threadIdx.x & 31;
    const float* src = x2 + (size_t)token * EMBED;
    float v0 = src[lane], v1 = src[lane + 32], v2 = src[lane + 64];
    float mu = wred_sum(v0 + v1 + v2) * (1.f / 96.f);
    float d0 = v0 - mu, d1 = v1 - mu, d2 = v2 - mu;
    float var = wred_sum(d0 * d0 + d1 * d1 + d2 * d2) * (1.f / 96.f);
    float rs = rsqrtf(var + 1e-5f);
    _Float16* dst = m16 + (size_t)token * EMBED;
    dst[lane]      = (_Float16)(d0 * rs * g[lane]      + b[lane]);
    dst[lane + 32] = (_Float16)(d1 * rs * g[lane + 32] + b[lane + 32]);
    dst[lane + 64] = (_Float16)(d2 * rs * g[lane + 64] + b[lane + 64]);
}

// ---------------- kernel 7: MLP1 (96 -> 384) + exact GELU ----------------
__global__ void k_mlp1(const _Float16* __restrict__ m16, const _Float16* __restrict__ wm1T,
                       const float* __restrict__ bm1, _Float16* __restrict__ h1) {
    int mt = blockIdx.x, z = blockIdx.y;
    int lane = threadIdx.x, l15 = lane & 15, hi = lane >> 4;
    v8f acc[6];
#pragma unroll
    for (int nt = 0; nt < 6; ++nt) acc[nt] = vzero();
    const _Float16* A = m16 + (size_t)mt * 16 * EMBED;
#pragma unroll
    for (int kt = 0; kt < 3; ++kt) {
        v16h a = load_a_frag(A + kt * 32, EMBED);
#pragma unroll
        for (int nt = 0; nt < 6; ++nt) {
            v16h bfr = load_b_frag(wm1T + (size_t)(z * 96 + nt * 16) * EMBED + kt * 32, EMBED);
            acc[nt] = wmma16(a, bfr, acc[nt]);
        }
    }
#pragma unroll
    for (int nt = 0; nt < 6; ++nt) {
        int c = z * 96 + nt * 16 + l15;
        float bias = bm1[c];
#pragma unroll
        for (int r = 0; r < 8; ++r) {
            int row = mt * 16 + r + 8 * hi;
            float v = acc[nt][r] + bias;
            float gl = 0.5f * v * (1.f + erff(v * 0.70710678118f));
            h1[(size_t)row * HIDDEN + c] = (_Float16)gl;
        }
    }
}

// ---------------- kernel 8: MLP2 (384 -> 96) + residual -> out ----------------
__global__ void k_mlp2(const _Float16* __restrict__ h1, const _Float16* __restrict__ wm2T,
                       const float* __restrict__ bm2, const float* __restrict__ x2,
                       float* __restrict__ out) {
    int mt = blockIdx.x;
    int lane = threadIdx.x, l15 = lane & 15, hi = lane >> 4;
    v8f acc[6];
#pragma unroll
    for (int nt = 0; nt < 6; ++nt) acc[nt] = vzero();
    const _Float16* A = h1 + (size_t)mt * 16 * HIDDEN;
#pragma unroll
    for (int kt = 0; kt < 12; ++kt) {
        v16h a = load_a_frag(A + kt * 32, HIDDEN);
#pragma unroll
        for (int nt = 0; nt < 6; ++nt) {
            v16h bfr = load_b_frag(wm2T + (size_t)(nt * 16) * HIDDEN + kt * 32, HIDDEN);
            acc[nt] = wmma16(a, bfr, acc[nt]);
        }
    }
#pragma unroll
    for (int nt = 0; nt < 6; ++nt) {
        int c = nt * 16 + l15;
        float bias = bm2[c];
#pragma unroll
        for (int r = 0; r < 8; ++r) {
            int row = mt * 16 + r + 8 * hi;
            size_t idx = (size_t)row * EMBED + c;
            out[idx] = x2[idx] + acc[nt][r] + bias;
        }
    }
}

// ---------------- host launch ----------------
extern "C" void kernel_launch(void* const* d_in, const int* in_sizes, int n_in,
                              void* d_out, int out_size, void* d_ws, size_t ws_size,
                              hipStream_t stream) {
    const float* x      = (const float*)d_in[0];
    const float* ln1_g  = (const float*)d_in[1];
    const float* ln1_b  = (const float*)d_in[2];
    const float* w_qkv  = (const float*)d_in[3];
    const float* b_qkv  = (const float*)d_in[4];
    const float* w_out  = (const float*)d_in[5];
    const float* b_out  = (const float*)d_in[6];
    const float* ln2_g  = (const float*)d_in[7];
    const float* ln2_b  = (const float*)d_in[8];
    const float* w_mlp1 = (const float*)d_in[9];
    const float* b_mlp1 = (const float*)d_in[10];
    const float* w_mlp2 = (const float*)d_in[11];
    const float* b_mlp2 = (const float*)d_in[12];

    char* ws = (char*)d_ws;
    _Float16* wqkvT = (_Float16*)(ws + OFF_WQKVT);
    _Float16* woutT = (_Float16*)(ws + OFF_WOUTT);
    _Float16* wm1T  = (_Float16*)(ws + OFF_WM1T);
    _Float16* wm2T  = (_Float16*)(ws + OFF_WM2T);
    _Float16* hw    = (_Float16*)(ws + OFF_HW);    // also reused as attention output o_buf
    _Float16* q     = (_Float16*)(ws + OFF_Q);
    _Float16* kk    = (_Float16*)(ws + OFF_K);
    _Float16* vT    = (_Float16*)(ws + OFF_VT);
    float*    x2    = (float*)(ws + OFF_X2);
    _Float16* h1    = (_Float16*)(ws + OFF_H1);
    _Float16* m16   = (_Float16*)(ws + OFF_M16);   // reuses q region (dead after attention)

    // zero hw + q + k + vT (padded rows/dims must be 0 so pads are inert)
    hipMemsetAsync(ws + OFF_HW, 0, OFF_X2 - OFF_HW, stream);

    k_cvt_weights<<<(36864 + 255) / 256, 256, 0, stream>>>(w_qkv, w_out, w_mlp1, w_mlp2,
                                                           wqkvT, woutT, wm1T, wm2T);
    k_ln1_partition<<<TOKENS / 8, 256, 0, stream>>>(x, ln1_g, ln1_b, hw);
    k_qkv<<<dim3(MT, NWINDOWS, 3), 32, 0, stream>>>(hw, wqkvT, b_qkv, q, kk, vT);
    k_attn<<<dim3(MT, NWH), 32, 0, stream>>>(q, kk, vT, hw /* o_buf */);
    k_proj<<<dim3(MT, NWINDOWS), 32, 0, stream>>>(hw, woutT, b_out, x, x2);
    k_ln2<<<TOKENS / 8, 256, 0, stream>>>(x2, ln2_g, ln2_b, m16);
    k_mlp1<<<dim3(TOKENS / 16, 4), 32, 0, stream>>>(m16, wm1T, b_mlp1, h1);
    k_mlp2<<<dim3(TOKENS / 16), 32, 0, stream>>>(h1, wm2T, b_mlp2, x2, (float*)d_out);
}
